// LSTMModel_80711025426545
// MI455X (gfx1250) — compile-verified, matrix operand order checked by reading, not run on hardware
//
#include <hip/hip_runtime.h>
#include <math.h>

// ---------------------------------------------------------------------------
// LSTM (2 layers, B=1024, SEQ=30, IN=HID=1028) via bf16 WMMA on gfx1250.
// Padded sizes: NP = 1088 (68 tiles of N=16), KP = 2*NP = 2176 (68 tiles of K=32).
// comb layout per row: [ x_pad (NP) | h_pad (NP) ]  (bf16 as ushort)
// Weights pre-swizzled to WMMA B-fragment order:
//   frag(g,kt,nt) = 512 contiguous bf16: lane L holds col N=nt*16+(L&15),
//   K = kt*32 + (L>>4)*16 + e  for e=0..15 (K ascending within lane).
// GEMM: block 256 = 8 wave32s, tile 128(M) x 64(N) per gate; A staged through
// double-buffered LDS; each wave owns 2 A-frags x 2 B-frags = 4 accumulators.
// ---------------------------------------------------------------------------

#define BATCH   1024
#define SEQ     30
#define INU     1028
#define HID     1028
#define NP      1088
#define KP      2176
#define KT      68     // K tiles of 32
#define NT      68     // N tiles of 16

#define BM      128            // rows per block
#define AROW    40             // LDS row stride in ushorts (80B, 16B-aligned, bank-spread)

typedef __attribute__((ext_vector_type(16))) __bf16 v16bf;
typedef __attribute__((ext_vector_type(8)))  float  v8f;

__device__ __forceinline__ unsigned short f2bf(float f) {
    unsigned int u = __float_as_uint(f);
    unsigned int r = u + 0x7FFFu + ((u >> 16) & 1u);   // round-to-nearest-even
    return (unsigned short)(r >> 16);
}

// ---------------------------------------------------------------------------
// Repack one layer's 4 gate weights [2056 x 1028] fp32 -> swizzled bf16 frags.
// Padded rows/cols are zero. Logical K: [0,1028)=x rows, [NP,NP+1028)=h rows.
// ---------------------------------------------------------------------------
__global__ __launch_bounds__(256)
void repack_weights(const float* __restrict__ Wf, const float* __restrict__ Wi,
                    const float* __restrict__ Wc, const float* __restrict__ Wo,
                    unsigned short* __restrict__ out)
{
    const long long total = 4LL * KT * NT * 512;
    long long idx = (long long)blockIdx.x * blockDim.x + threadIdx.x;
    if (idx >= total) return;

    int e    = (int)(idx & 15);
    int lane = (int)((idx >> 4) & 31);
    long long frag = idx >> 9;          // (g*KT + kt)*NT + nt
    int nt = (int)(frag % NT);
    int kt = (int)((frag / NT) % KT);
    int g  = (int)(frag / (NT * KT));

    int k    = kt * 32 + ((lane >> 4) << 4) + e;   // padded K index
    int ncol = nt * 16 + (lane & 15);              // padded N index

    int srcRow = -1;
    if (k < INU)                      srcRow = k;                 // x part
    else if (k >= NP && k < NP + HID) srcRow = INU + (k - NP);    // h part

    float v = 0.0f;
    if (srcRow >= 0 && ncol < HID) {
        const float* W = (g == 0) ? Wf : (g == 1) ? Wi : (g == 2) ? Wc : Wo;
        v = W[(long long)srcRow * HID + ncol];
    }
    out[idx] = f2bf(v);
}

__global__ void prep_bias(const float* __restrict__ bf, const float* __restrict__ bi,
                          const float* __restrict__ bc, const float* __restrict__ bo,
                          float* __restrict__ out)
{
    int idx = blockIdx.x * blockDim.x + threadIdx.x;
    if (idx >= 4 * NP) return;
    int g = idx / NP, n = idx % NP;
    const float* s = (g == 0) ? bf : (g == 1) ? bi : (g == 2) ? bc : bo;
    out[idx] = (n < HID) ? s[n] : 0.0f;
}

__global__ void init_state(const float* __restrict__ h1, const float* __restrict__ c1,
                           const float* __restrict__ h2, const float* __restrict__ c2,
                           unsigned short* __restrict__ comb1, unsigned short* __restrict__ comb2,
                           float* __restrict__ c1s, float* __restrict__ c2s,
                           float* __restrict__ h2f)
{
    long long idx = (long long)blockIdx.x * blockDim.x + threadIdx.x;
    if (idx >= (long long)BATCH * NP) return;
    int m = (int)(idx / NP), n = (int)(idx % NP);
    bool in = n < HID;
    float vh1 = in ? h1[(long long)m * HID + n] : 0.0f;
    float vh2 = in ? h2[(long long)m * HID + n] : 0.0f;
    float vc1 = in ? c1[(long long)m * HID + n] : 0.0f;
    float vc2 = in ? c2[(long long)m * HID + n] : 0.0f;
    comb1[(long long)m * KP + n]      = 0;          // x slot (rewritten each step)
    comb1[(long long)m * KP + NP + n] = f2bf(vh1);  // recurrent h1
    comb2[(long long)m * KP + n]      = 0;          // layer2 input slot
    comb2[(long long)m * KP + NP + n] = f2bf(vh2);  // recurrent h2
    c1s[idx] = vc1;
    c2s[idx] = vc2;
    h2f[idx] = vh2;
}

__global__ void load_x(const float* __restrict__ x, unsigned short* __restrict__ comb1, int t)
{
    long long idx = (long long)blockIdx.x * blockDim.x + threadIdx.x;
    if (idx >= (long long)BATCH * NP) return;
    int m = (int)(idx / NP), n = (int)(idx % NP);
    float v = (n < INU) ? x[((long long)m * SEQ + t) * INU + n] : 0.0f;
    comb1[(long long)m * KP + n] = f2bf(v);
}

// ---------------------------------------------------------------------------
// Gate GEMM: gates[g][m][n] = sum_k comb[m][k] * W_g[k][n]   (no bias here)
// grid = (NP/64, BATCH/BM, 4 gates), block = 256 (8 wave32s).
// Wave w: row tiles 2*(w>>1)+{0,1}, col tiles 2*(w&1)+{0,1} -> 4 accumulators.
// A tile [BM x 32] staged through double-buffered LDS (one barrier per K-tile).
// ---------------------------------------------------------------------------
__global__ __launch_bounds__(256)
void lstm_gemm(const unsigned short* __restrict__ comb,
               const unsigned short* __restrict__ wsw,
               float* __restrict__ gates)
{
    __shared__ unsigned short sA[2][BM * AROW];   // 2 x 10 KB, padded rows

    const int lane = threadIdx.x & 31;
    const int wave = threadIdx.x >> 5;
    const int g    = blockIdx.z;

    const int m0  = blockIdx.y * BM;
    const int rt0 = (wave >> 1) * 2;              // first of two row tiles (0..7)
    const int ct0 = (wave & 1) * 2;               // first of two col tiles (0..3)

    // cooperative A staging: thread i copies 32B = row (i>>1), half (i&1)
    const int srow  = threadIdx.x >> 1;
    const int shalf = threadIdx.x & 1;
    const unsigned short* sSrc = comb + (long long)(m0 + srow) * KP + shalf * 16;
    unsigned short* sDst0 = &sA[0][srow * AROW + shalf * 16];
    unsigned short* sDst1 = &sA[1][srow * AROW + shalf * 16];

    // LDS read addresses for this wave's two A fragments
    const unsigned short* aRd0a = &sA[0][(rt0 * 16 + (lane & 15)) * AROW + ((lane >> 4) << 4)];
    const unsigned short* aRd0b = aRd0a + 16 * AROW;
    const unsigned short* aRd1a = &sA[1][(rt0 * 16 + (lane & 15)) * AROW + ((lane >> 4) << 4)];
    const unsigned short* aRd1b = aRd1a + 16 * AROW;

    // B fragments: swizzled, lane-contiguous
    const long long ktStride = (long long)NT * 512;
    const unsigned short* bBase =
        wsw + ((((long long)g * KT) * NT + blockIdx.x * 4 + ct0) * 32 + lane) * 16;

    v8f acc00 = {}, acc01 = {}, acc10 = {}, acc11 = {};

    // prologue: stage kt=0 into buffer 0
    {
        const unsigned short* s = sSrc;
        *(uint4*)sDst0       = *(const uint4*)s;
        *(uint4*)(sDst0 + 8) = *(const uint4*)(s + 8);
    }
    __syncthreads();

    for (int kt = 0; kt < KT; ++kt) {
        const int buf = kt & 1;

        // stage next K-tile into the other buffer while we compute this one
        if (kt + 1 < KT) {
            const unsigned short* s = sSrc + (kt + 1) * 32;
            unsigned short* d = buf ? sDst0 : sDst1;
            *(uint4*)d       = *(const uint4*)s;
            *(uint4*)(d + 8) = *(const uint4*)(s + 8);
        }

        const unsigned short* bp = bBase + (long long)kt * ktStride;
        v16bf b0 = *(const v16bf*)(bp);
        v16bf b1 = *(const v16bf*)(bp + 512);
        __builtin_prefetch(bp + 2 * ktStride, 0, 1);   // global_prefetch_b8

        v16bf a0 = *(const v16bf*)(buf ? aRd1a : aRd0a);
        v16bf a1 = *(const v16bf*)(buf ? aRd1b : aRd0b);

        acc00 = __builtin_amdgcn_wmma_f32_16x16x32_bf16(false, a0, false, b0, (short)0, acc00, false, false);
        acc01 = __builtin_amdgcn_wmma_f32_16x16x32_bf16(false, a0, false, b1, (short)0, acc01, false, false);
        acc10 = __builtin_amdgcn_wmma_f32_16x16x32_bf16(false, a1, false, b0, (short)0, acc10, false, false);
        acc11 = __builtin_amdgcn_wmma_f32_16x16x32_bf16(false, a1, false, b1, (short)0, acc11, false, false);

        __syncthreads();   // stage stores visible; buffer safe to overwrite next iter
    }

    // C/D layout: VGPR v -> M = mbase + v + (lane>=16 ? 8 : 0), N = nbase + (lane&15)
    const int mA   = m0 + rt0 * 16 + ((lane >> 4) << 3);   // rows for acc0x
    const int mB   = mA + 16;                              // rows for acc1x
    const int colA = blockIdx.x * 64 + ct0 * 16 + (lane & 15);
    float* gbase = gates + (long long)g * BATCH * NP;
#pragma unroll
    for (int v = 0; v < 8; ++v) {
        gbase[(long long)(mA + v) * NP + colA]      = acc00[v];
        gbase[(long long)(mA + v) * NP + colA + 16] = acc01[v];
        gbase[(long long)(mB + v) * NP + colA]      = acc10[v];
        gbase[(long long)(mB + v) * NP + colA + 16] = acc11[v];
    }
}

// ---------------------------------------------------------------------------
// LSTM cell + ReLU. Writes bf16 h into recurrent slot (hrec, col NP+n) and
// optionally next layer's input slot (hnext, col n) and fp32 copy (hf32).
// Padded columns (n>=1028): bias=0, weights=0 -> g=tanh(0)=0, c_old=0 -> c=0,
// h=o*tanh(0)=0, so pads remain exactly zero with no masking.
// ---------------------------------------------------------------------------
__global__ void lstm_cell(const float* __restrict__ gates, const float* __restrict__ bias,
                          float* __restrict__ cstate,
                          unsigned short* __restrict__ hrec,
                          unsigned short* __restrict__ hnext,
                          float* __restrict__ hf32)
{
    long long idx = (long long)blockIdx.x * blockDim.x + threadIdx.x;
    if (idx >= (long long)BATCH * NP) return;
    int m = (int)(idx / NP), n = (int)(idx % NP);
    const long long G = (long long)BATCH * NP;
    long long off = (long long)m * NP + n;

    float pf = gates[off + 0 * G] + bias[0 * NP + n];
    float pi = gates[off + 1 * G] + bias[1 * NP + n];
    float pg = gates[off + 2 * G] + bias[2 * NP + n];
    float po = gates[off + 3 * G] + bias[3 * NP + n];

    float f  = 1.0f / (1.0f + __expf(-pf));
    float i  = 1.0f / (1.0f + __expf(-pi));
    float gg = tanhf(pg);
    float o  = 1.0f / (1.0f + __expf(-po));

    float c = f * cstate[off] + i * gg;
    float h = o * tanhf(c);
    h = fmaxf(h, 0.0f);   // ReLU (fed back into recurrence, per reference)

    cstate[off] = c;
    unsigned short hb = f2bf(h);
    hrec[(long long)m * KP + NP + n] = hb;
    if (hnext) hnext[(long long)m * KP + n] = hb;
    if (hf32)  hf32[off] = h;
}

// one block per batch row; strided partial sums + LDS tree reduction
__global__ __launch_bounds__(256)
void fc_out(const float* __restrict__ h2f, const float* __restrict__ Wfc,
            const float* __restrict__ bfc, float* __restrict__ out)
{
    __shared__ float red[256];
    const int m = blockIdx.x;
    const float* hr = h2f + (long long)m * NP;
    float s = 0.0f;
    for (int n = threadIdx.x; n < HID; n += 256) s += hr[n] * Wfc[n];
    red[threadIdx.x] = s;
    __syncthreads();
#pragma unroll
    for (int st = 128; st > 0; st >>= 1) {
        if ((int)threadIdx.x < st) red[threadIdx.x] += red[threadIdx.x + st];
        __syncthreads();
    }
    if (threadIdx.x == 0) out[m] = red[0] + bfc[0];
}

// ---------------------------------------------------------------------------
// Host driver.  Input order (flattened dict): x,h1,c1,h2,c2,
//   p1{Wf,bf,Wi,bi,Wc,bc,Wo,bo}, p2{...}, Wfc, bfc
// ---------------------------------------------------------------------------
extern "C" void kernel_launch(void* const* d_in, const int* in_sizes, int n_in,
                              void* d_out, int out_size, void* d_ws, size_t ws_size,
                              hipStream_t stream)
{
    const float* x   = (const float*)d_in[0];
    const float* h1  = (const float*)d_in[1];
    const float* c1  = (const float*)d_in[2];
    const float* h2  = (const float*)d_in[3];
    const float* c2  = (const float*)d_in[4];
    const float* W1f = (const float*)d_in[5];
    const float* b1f = (const float*)d_in[6];
    const float* W1i = (const float*)d_in[7];
    const float* b1i = (const float*)d_in[8];
    const float* W1c = (const float*)d_in[9];
    const float* b1c = (const float*)d_in[10];
    const float* W1o = (const float*)d_in[11];
    const float* b1o = (const float*)d_in[12];
    const float* W2f = (const float*)d_in[13];
    const float* b2f = (const float*)d_in[14];
    const float* W2i = (const float*)d_in[15];
    const float* b2i = (const float*)d_in[16];
    const float* W2c = (const float*)d_in[17];
    const float* b2c = (const float*)d_in[18];
    const float* W2o = (const float*)d_in[19];
    const float* b2o = (const float*)d_in[20];
    const float* Wfc = (const float*)d_in[21];
    const float* bfc = (const float*)d_in[22];
    float* out = (float*)d_out;

    // workspace carve (256B aligned)
    char* p = (char*)d_ws;
    auto carve = [&](size_t bytes) -> char* {
        char* r = p;
        p += (bytes + 255) & ~(size_t)255;
        return r;
    };
    const size_t wswElems = 4ULL * KT * NT * 512;
    unsigned short* wsw1  = (unsigned short*)carve(wswElems * 2);
    unsigned short* wsw2  = (unsigned short*)carve(wswElems * 2);
    unsigned short* comb1 = (unsigned short*)carve((size_t)BATCH * KP * 2);
    unsigned short* comb2 = (unsigned short*)carve((size_t)BATCH * KP * 2);
    float* gatesBuf = (float*)carve(4ULL * BATCH * NP * 4);
    float* c1s  = (float*)carve((size_t)BATCH * NP * 4);
    float* c2s  = (float*)carve((size_t)BATCH * NP * 4);
    float* h2f  = (float*)carve((size_t)BATCH * NP * 4);
    float* bias1 = (float*)carve(4ULL * NP * 4);
    float* bias2 = (float*)carve(4ULL * NP * 4);

    // one-time setup (deterministic; runs every call)
    {
        long long total = (long long)wswElems;
        int blocks = (int)((total + 255) / 256);
        repack_weights<<<blocks, 256, 0, stream>>>(W1f, W1i, W1c, W1o, wsw1);
        repack_weights<<<blocks, 256, 0, stream>>>(W2f, W2i, W2c, W2o, wsw2);
    }
    prep_bias<<<(4 * NP + 255) / 256, 256, 0, stream>>>(b1f, b1i, b1c, b1o, bias1);
    prep_bias<<<(4 * NP + 255) / 256, 256, 0, stream>>>(b2f, b2i, b2c, b2o, bias2);

    const long long elemN = (long long)BATCH * NP;
    const int eBlocks = (int)((elemN + 255) / 256);
    init_state<<<eBlocks, 256, 0, stream>>>(h1, c1, h2, c2, comb1, comb2, c1s, c2s, h2f);

    dim3 gemmGrid(NP / 64, BATCH / BM, 4);
    for (int t = 0; t < SEQ; ++t) {
        load_x<<<eBlocks, 256, 0, stream>>>(x, comb1, t);
        lstm_gemm<<<gemmGrid, 256, 0, stream>>>(comb1, wsw1, gatesBuf);
        lstm_cell<<<eBlocks, 256, 0, stream>>>(gatesBuf, bias1, c1s, comb1, comb2, nullptr);
        lstm_gemm<<<gemmGrid, 256, 0, stream>>>(comb2, wsw2, gatesBuf);
        lstm_cell<<<eBlocks, 256, 0, stream>>>(gatesBuf, bias2, c2s, comb2, nullptr, h2f);
    }
    fc_out<<<1024, 256, 0, stream>>>(h2f, Wfc, bfc, out);
}
